// RNN_model_6098853561018
// MI455X (gfx1250) — matrix operand order repaired and checked by reading, not compiled, not run on hardware
//
#include <hip/hip_runtime.h>
#include <hip/hip_bf16.h>

typedef __attribute__((ext_vector_type(16))) _Float16 v16h;
typedef __attribute__((ext_vector_type(8)))  float    v8f;

// ---------------- problem constants ----------------
#define B_SZ  2048
#define T_SZ  200
#define XDIM  65      // 64 feats + click
// hidden U = 64, 4U = 256, P0 = 64, P1 = 32, NC = 1

// ---------------- LDS layout ----------------
#define PW 136        // pitch (halfs) for K=128 operands (128 + 8 pad)
#define PS 72         // pitch (halfs) for K=64 operands  (64 + 8 pad)

constexpr int OFF_WCT  = 0;                       // [Wxc;Whc]^T : 256 x PW halfs
constexpr int OFF_WVT  = OFF_WCT  + 256*PW*2;     // [Wxv;Whv]^T
constexpr int OFF_WSCC = OFF_WVT  + 256*PW*2;     // 64 x PS halfs each
constexpr int OFF_WSCV = OFF_WSCC + 64*PS*2;
constexpr int OFF_WSVV = OFF_WSCV + 64*PS*2;
constexpr int OFF_WSVC = OFF_WSVV + 64*PS*2;
constexpr int OFF_WPC0 = OFF_WSVC + 64*PS*2;
constexpr int OFF_WPV0 = OFF_WPC0 + 64*PS*2;
constexpr int OFF_WPC1 = OFF_WPV0 + 64*PS*2;      // 32 x PS halfs
constexpr int OFF_WPV1 = OFF_WPC1 + 32*PS*2;
constexpr int OFF_BIAS = OFF_WPV1 + 32*PS*2;      // float region (biases + Wfcc/Wfcv)
constexpr int OFF_STAGE= OFF_BIAS + 4096;

// per-wave staging
constexpr int STG_AC = 0;                 // A = [feats | H_c] : 16 x PW halfs
constexpr int STG_AV = STG_AC + 16*PW*2;  // A = [feats | H_v]
constexpr int STG_H1 = STG_AV + 16*PW*2;  // MLP hidden (16 x PS halfs)
constexpr int STG_P2 = STG_H1 + 16*PS*2;  // MLP layer-2 out, 16x32 f32
constexpr int STG_SC = STG_P2 + 16*32*4;  // scratch floats: gprev[16] gcur[16] pc[16]
constexpr int STG_BYTES = STG_SC + 256;   // 13312 B
constexpr int NWAVES = 4;
constexpr int SMEM_BYTES = OFF_STAGE + NWAVES * STG_BYTES;   // 261120 B < 320 KB

// bias float indices
#define BI_BXC  0
#define BI_BXV  256
#define BI_BPC0 512
#define BI_BPC1 576
#define BI_BPV0 608
#define BI_BPV1 672
#define BI_WFCC 704
#define BI_WFCV 736
#define BI_BFCC 768
#define BI_BFCV 769

// ---------------- helpers ----------------
__device__ __forceinline__ void wave_lds_fence() {
  __builtin_amdgcn_wave_barrier();
  asm volatile("s_wait_dscnt 0x0" ::: "memory");
  __builtin_amdgcn_wave_barrier();
}
__device__ __forceinline__ float sigmoidf_(float x) { return 1.0f / (1.0f + __expf(-x)); }
__device__ __forceinline__ float leakyf_(float x)   { return x > 0.0f ? x : 0.3f * x; }
__device__ __forceinline__ v8f zero8() {
  v8f r;
#pragma unroll
  for (int i = 0; i < 8; ++i) r[i] = 0.0f;
  return r;
}

// A fragment (16xK row-major staging in LDS), per ISA 16-bit A 16x32 layout:
// lane<16 : row=lane, halves K = k0+0..7 and k0+16..23
// lane>=16: row=lane-16, halves K = k0+8..15 and k0+24..31
__device__ __forceinline__ v16h load_A(const _Float16* a, int pitch, int k0, int lane) {
  const _Float16* p = a + (lane & 15) * pitch + k0 + ((lane & 16) ? 8 : 0);
  v16h r;
#pragma unroll
  for (int i = 0; i < 8; ++i) { r[i] = p[i]; r[8 + i] = p[16 + i]; }
  return r;
}

// B fragment from transposed weights W^T[n][k] (pitch halfs), per ISA B 32x16 layout:
// lane<16 : col n=n0+lane, K = k0+0..7 and k0+16..23 ; lane>=16: col n=n0+lane-16, K = k0+8..15 and k0+24..31
__device__ __forceinline__ v16h load_B(const _Float16* w, int pitch, int n0, int k0, int lane) {
  const _Float16* p = w + (n0 + (lane & 15)) * pitch + k0 + ((lane & 16) ? 8 : 0);
  v16h r;
#pragma unroll
  for (int i = 0; i < 8; ++i) { r[i] = p[i]; r[8 + i] = p[16 + i]; }
  return r;
}

__device__ __forceinline__ v8f WMMA(v16h a, v16h b, v8f c) {
  return __builtin_amdgcn_wmma_f32_16x16x32_f16(false, a, false, b, (short)0, c, false, false);
}

// ---------------- persistent RNN kernel ----------------
__global__ __launch_bounds__(128, 1)
void rnn_persistent_kernel(
    const float* __restrict__ x,
    const float* __restrict__ Wxc, const float* __restrict__ bxc, const float* __restrict__ Whc,
    const float* __restrict__ Wxv, const float* __restrict__ bxv, const float* __restrict__ Whv,
    const float* __restrict__ Wscc, const float* __restrict__ Wscv,
    const float* __restrict__ Wsvv, const float* __restrict__ Wsvc,
    const float* __restrict__ Wpc0, const float* __restrict__ bpc0,
    const float* __restrict__ Wpc1, const float* __restrict__ bpc1,
    const float* __restrict__ Wfcc, const float* __restrict__ bfcc,
    const float* __restrict__ Wpv0, const float* __restrict__ bpv0,
    const float* __restrict__ Wpv1, const float* __restrict__ bpv1,
    const float* __restrict__ Wfcv, const float* __restrict__ bfcv,
    float* __restrict__ out)
{
  extern __shared__ char smem[];
  _Float16* WcT   = (_Float16*)(smem + OFF_WCT);
  _Float16* WvT   = (_Float16*)(smem + OFF_WVT);
  _Float16* WsccT = (_Float16*)(smem + OFF_WSCC);
  _Float16* WscvT = (_Float16*)(smem + OFF_WSCV);
  _Float16* WsvvT = (_Float16*)(smem + OFF_WSVV);
  _Float16* WsvcT = (_Float16*)(smem + OFF_WSVC);
  _Float16* Wpc0T = (_Float16*)(smem + OFF_WPC0);
  _Float16* Wpv0T = (_Float16*)(smem + OFF_WPV0);
  _Float16* Wpc1T = (_Float16*)(smem + OFF_WPC1);
  _Float16* Wpv1T = (_Float16*)(smem + OFF_WPV1);
  float*    biasF = (float*)  (smem + OFF_BIAS);

  const int tid  = threadIdx.x;
  const int lane = tid & 31;
  const int wave = tid >> 5;

  // ---- cooperative weight load: transpose + fp32->fp16 ----
  for (int i = tid; i < 256 * 128; i += 128) {
    int n = i >> 7, k = i & 127;
    WcT[n * PW + k] = (_Float16)((k < 64) ? Wxc[k * 256 + n] : Whc[(k - 64) * 256 + n]);
    WvT[n * PW + k] = (_Float16)((k < 64) ? Wxv[k * 256 + n] : Whv[(k - 64) * 256 + n]);
  }
  for (int i = tid; i < 64 * 64; i += 128) {
    int n = i >> 6, k = i & 63;
    WsccT[n * PS + k] = (_Float16)Wscc[k * 64 + n];
    WscvT[n * PS + k] = (_Float16)Wscv[k * 64 + n];
    WsvvT[n * PS + k] = (_Float16)Wsvv[k * 64 + n];
    WsvcT[n * PS + k] = (_Float16)Wsvc[k * 64 + n];
    Wpc0T[n * PS + k] = (_Float16)Wpc0[k * 64 + n];
    Wpv0T[n * PS + k] = (_Float16)Wpv0[k * 64 + n];
  }
  for (int i = tid; i < 32 * 64; i += 128) {
    int n = i >> 6, k = i & 63;
    Wpc1T[n * PS + k] = (_Float16)Wpc1[k * 32 + n];
    Wpv1T[n * PS + k] = (_Float16)Wpv1[k * 32 + n];
  }
  for (int i = tid; i < 256; i += 128) { biasF[BI_BXC + i] = bxc[i]; biasF[BI_BXV + i] = bxv[i]; }
  for (int i = tid; i < 64;  i += 128) { biasF[BI_BPC0 + i] = bpc0[i]; biasF[BI_BPV0 + i] = bpv0[i]; }
  for (int i = tid; i < 32;  i += 128) {
    biasF[BI_BPC1 + i] = bpc1[i]; biasF[BI_BPV1 + i] = bpv1[i];
    biasF[BI_WFCC + i] = Wfcc[i]; biasF[BI_WFCV + i] = Wfcv[i];
  }
  if (tid == 0) { biasF[BI_BFCC] = bfcc[0]; biasF[BI_BFCV] = bfcv[0]; }
  __syncthreads();

  // ---- per-wave staging pointers ----
  char* stg = smem + OFF_STAGE + wave * STG_BYTES;
  _Float16* Ac = (_Float16*)(stg + STG_AC);   // [feats(0:64) | H_c(64:128)]
  _Float16* Av = (_Float16*)(stg + STG_AV);   // [feats(0:64) | H_v(64:128)]
  _Float16* H1 = (_Float16*)(stg + STG_H1);
  float*    P2 = (float*)(stg + STG_P2);
  float*    scr = (float*)(stg + STG_SC);
  float* gprev = scr;        // [16]
  float* gcur  = scr + 16;   // [16]
  float* pcr   = scr + 32;   // [16]

  const int b0    = blockIdx.x * 64 + wave * 16;   // 16 batch rows per wave
  const int nlane = lane & 15;
  const int moff  = (lane & 16) ? 8 : 0;

  // ---- init state ----
  for (int i = lane; i < 16 * 64; i += 32) {
    int r = i >> 6, u = i & 63;
    Ac[r * PW + 64 + u] = (_Float16)0.0f;
    Av[r * PW + 64 + u] = (_Float16)0.0f;
  }
  if (lane < 16) gprev[lane] = 0.0f;
  v8f s_c[4], s_v[4];
#pragma unroll
  for (int jt = 0; jt < 4; ++jt) { s_c[jt] = zero8(); s_v[jt] = zero8(); }
  wave_lds_fence();

  // ================= time loop (sequential dependency) =================
#pragma unroll 1
  for (int t = 0; t < T_SZ; ++t) {
    // ---- stage input features (fp32 -> fp16, into both A buffers) ----
    for (int i = lane; i < 16 * 64; i += 32) {
      int r = i >> 6, f = i & 63;
      float v = x[((size_t)(b0 + r) * T_SZ + t) * XDIM + f];
      _Float16 h = (_Float16)v;
      Ac[r * PW + f] = h;
      Av[r * PW + f] = h;
    }
    if (lane < 16) {
      float cl = x[((size_t)(b0 + lane) * T_SZ + t) * XDIM + 64];
      gcur[lane] = (cl >= 0.5f) ? 1.0f : 0.0f;
      if (t + 1 < T_SZ)
        __builtin_prefetch(&x[((size_t)(b0 + lane) * T_SZ + (t + 1)) * XDIM], 0, 1);
    }
    wave_lds_fence();

    // ================= c side =================
    v8f hcn[4];
    for (int jt = 0; jt < 4; ++jt) {
      // zero accumulators -> first WMMA folds C to inline 0 (no init movs);
      // per-column bias folded into the activation epilogue instead.
      v8f aF = zero8(), aI = zero8(), aO = zero8(), aG = zero8();
#pragma unroll
      for (int kc = 0; kc < 4; ++kc) {            // fused [feats|H_c] @ [Wxc;Whc], K=128
        v16h a = load_A(Ac, PW, kc * 32, lane);
        aF = WMMA(a, load_B(WcT, PW,       jt * 16, kc * 32, lane), aF);
        aI = WMMA(a, load_B(WcT, PW,  64 + jt * 16, kc * 32, lane), aI);
        aO = WMMA(a, load_B(WcT, PW, 128 + jt * 16, kc * 32, lane), aO);
        aG = WMMA(a, load_B(WcT, PW, 192 + jt * 16, kc * 32, lane), aG);
      }
      v8f aCC = zero8(), aCV = zero8();
#pragma unroll
      for (int kc = 0; kc < 2; ++kc) {            // H_c@Wscc, H_v@Wscv (K=64)
        v16h ahc = load_A(Ac, PW, 64 + kc * 32, lane);
        aCC = WMMA(ahc, load_B(WsccT, PS, jt * 16, kc * 32, lane), aCC);
        v16h ahv = load_A(Av, PW, 64 + kc * 32, lane);
        aCV = WMMA(ahv, load_B(WscvT, PS, jt * 16, kc * 32, lane), aCV);
      }
      float bF = biasF[BI_BXC +       jt * 16 + nlane];
      float bI = biasF[BI_BXC +  64 + jt * 16 + nlane];
      float bO = biasF[BI_BXC + 128 + jt * 16 + nlane];
      float bG = biasF[BI_BXC + 192 + jt * 16 + nlane];
#pragma unroll
      for (int r = 0; r < 8; ++r) {
        float gp = gprev[r + moff];
        float fg = sigmoidf_(aF[r] + bF);
        float ig = sigmoidf_(aI[r] + bI);
        float og = sigmoidf_(aO[r] + bO);
        float gg = tanhf(aG[r] + bG);
        float sh = tanhf((1.0f - gp) * aCC[r] + gp * aCV[r]);
        float sc = sh + ig * gg + (1.0f - gp) * (fg * s_c[jt][r]);
        s_c[jt][r] = sc;
        hcn[jt][r] = og * tanhf(sc);
      }
    }
    wave_lds_fence();
#pragma unroll
    for (int jt = 0; jt < 4; ++jt)                // stage new H_c (A operand for Wsvc/MLP/next-t)
#pragma unroll
      for (int r = 0; r < 8; ++r)
        Ac[(r + moff) * PW + 64 + jt * 16 + nlane] = (_Float16)hcn[jt][r];
    wave_lds_fence();

    // ---- c prediction head ----
    for (int jt = 0; jt < 4; ++jt) {
      v8f acc = zero8();
#pragma unroll
      for (int kc = 0; kc < 2; ++kc) {
        v16h a = load_A(Ac, PW, 64 + kc * 32, lane);
        acc = WMMA(a, load_B(Wpc0T, PS, jt * 16, kc * 32, lane), acc);
      }
      float b0f = biasF[BI_BPC0 + jt * 16 + nlane];
#pragma unroll
      for (int r = 0; r < 8; ++r)
        H1[(r + moff) * PS + jt * 16 + nlane] = (_Float16)leakyf_(acc[r] + b0f);
    }
    wave_lds_fence();
    for (int jt = 0; jt < 2; ++jt) {
      v8f acc = zero8();
#pragma unroll
      for (int kc = 0; kc < 2; ++kc) {
        v16h a = load_A(H1, PS, kc * 32, lane);
        acc = WMMA(a, load_B(Wpc1T, PS, jt * 16, kc * 32, lane), acc);
      }
      float b1f = biasF[BI_BPC1 + jt * 16 + nlane];
#pragma unroll
      for (int r = 0; r < 8; ++r)
        P2[(r + moff) * 32 + jt * 16 + nlane] = leakyf_(acc[r] + b1f);
    }
    wave_lds_fence();
    if (lane < 16) {                               // 32 -> 1 + sigmoid, one row per lane
      float d = biasF[BI_BFCC];
      for (int j = 0; j < 32; ++j) d += P2[lane * 32 + j] * biasF[BI_WFCC + j];
      float p = sigmoidf_(d);
      pcr[lane] = p;
      out[(((size_t)(b0 + lane)) * T_SZ + t) * 2 + 0] = p;
    }
    wave_lds_fence();

    // ================= v side =================
    v8f hvn[4];
    for (int jt = 0; jt < 4; ++jt) {
      v8f aF = zero8(), aI = zero8(), aO = zero8(), aG = zero8();
#pragma unroll
      for (int kc = 0; kc < 4; ++kc) {            // fused [feats|H_v] @ [Wxv;Whv]
        v16h a = load_A(Av, PW, kc * 32, lane);
        aF = WMMA(a, load_B(WvT, PW,       jt * 16, kc * 32, lane), aF);
        aI = WMMA(a, load_B(WvT, PW,  64 + jt * 16, kc * 32, lane), aI);
        aO = WMMA(a, load_B(WvT, PW, 128 + jt * 16, kc * 32, lane), aO);
        aG = WMMA(a, load_B(WvT, PW, 192 + jt * 16, kc * 32, lane), aG);
      }
      v8f aVV = zero8(), aVC = zero8();
#pragma unroll
      for (int kc = 0; kc < 2; ++kc) {            // H_v@Wsvv, (new)H_c@Wsvc
        v16h ahv = load_A(Av, PW, 64 + kc * 32, lane);
        aVV = WMMA(ahv, load_B(WsvvT, PS, jt * 16, kc * 32, lane), aVV);
        v16h ahc = load_A(Ac, PW, 64 + kc * 32, lane);
        aVC = WMMA(ahc, load_B(WsvcT, PS, jt * 16, kc * 32, lane), aVC);
      }
      float bF = biasF[BI_BXV +       jt * 16 + nlane];
      float bI = biasF[BI_BXV +  64 + jt * 16 + nlane];
      float bO = biasF[BI_BXV + 128 + jt * 16 + nlane];
      float bG = biasF[BI_BXV + 192 + jt * 16 + nlane];
#pragma unroll
      for (int r = 0; r < 8; ++r) {
        int m = r + moff;
        float g  = gcur[m];
        float fg = sigmoidf_(aF[r] + bF);
        float ig = sigmoidf_(aI[r] + bI);
        float og = sigmoidf_(aO[r] + bO);
        float gg = tanhf(aG[r] + bG);
        float svh = tanhf(aVV[r] + g * aVC[r]);
        float svo = s_v[jt][r];
        float sv  = svh + (1.0f - g) * svo + g * (fg * svo + ig * gg);
        s_v[jt][r] = sv;
        float hvo = (float)Av[m * PW + 64 + jt * 16 + nlane];
        hvn[jt][r] = (1.0f - g) * hvo + g * (og * tanhf(sv));
      }
    }
    wave_lds_fence();
#pragma unroll
    for (int jt = 0; jt < 4; ++jt)
#pragma unroll
      for (int r = 0; r < 8; ++r)
        Av[(r + moff) * PW + 64 + jt * 16 + nlane] = (_Float16)hvn[jt][r];
    wave_lds_fence();

    // ---- v prediction head ----
    for (int jt = 0; jt < 4; ++jt) {
      v8f acc = zero8();
#pragma unroll
      for (int kc = 0; kc < 2; ++kc) {
        v16h a = load_A(Av, PW, 64 + kc * 32, lane);
        acc = WMMA(a, load_B(Wpv0T, PS, jt * 16, kc * 32, lane), acc);
      }
      float b0f = biasF[BI_BPV0 + jt * 16 + nlane];
#pragma unroll
      for (int r = 0; r < 8; ++r)
        H1[(r + moff) * PS + jt * 16 + nlane] = (_Float16)leakyf_(acc[r] + b0f);
    }
    wave_lds_fence();
    for (int jt = 0; jt < 2; ++jt) {
      v8f acc = zero8();
#pragma unroll
      for (int kc = 0; kc < 2; ++kc) {
        v16h a = load_A(H1, PS, kc * 32, lane);
        acc = WMMA(a, load_B(Wpv1T, PS, jt * 16, kc * 32, lane), acc);
      }
      float b1f = biasF[BI_BPV1 + jt * 16 + nlane];
#pragma unroll
      for (int r = 0; r < 8; ++r)
        P2[(r + moff) * 32 + jt * 16 + nlane] = leakyf_(acc[r] + b1f);
    }
    wave_lds_fence();
    if (lane < 16) {
      float d = biasF[BI_BFCV];
      for (int j = 0; j < 32; ++j) d += P2[lane * 32 + j] * biasF[BI_WFCV + j];
      float p = sigmoidf_(d) * pcr[lane];
      out[(((size_t)(b0 + lane)) * T_SZ + t) * 2 + 1] = p;
    }
    wave_lds_fence();
    if (lane < 16) gprev[lane] = gcur[lane];       // carry g = (click_t >= 0.5)
    wave_lds_fence();
  }
}

// ---------------- host launch ----------------
extern "C" void kernel_launch(void* const* d_in, const int* in_sizes, int n_in,
                              void* d_out, int out_size, void* d_ws, size_t ws_size,
                              hipStream_t stream) {
  (void)in_sizes; (void)n_in; (void)out_size; (void)d_ws; (void)ws_size;
  const float* x    = (const float*)d_in[0];
  const float* Wxc  = (const float*)d_in[1];
  const float* bxc  = (const float*)d_in[2];
  const float* Whc  = (const float*)d_in[3];
  const float* Wxv  = (const float*)d_in[4];
  const float* bxv  = (const float*)d_in[5];
  const float* Whv  = (const float*)d_in[6];
  const float* Wscc = (const float*)d_in[7];
  const float* Wscv = (const float*)d_in[8];
  const float* Wsvv = (const float*)d_in[9];
  const float* Wsvc = (const float*)d_in[10];
  const float* Wpc0 = (const float*)d_in[11];
  const float* bpc0 = (const float*)d_in[12];
  const float* Wpc1 = (const float*)d_in[13];
  const float* bpc1 = (const float*)d_in[14];
  const float* Wfcc = (const float*)d_in[15];
  const float* bfcc = (const float*)d_in[16];
  const float* Wpv0 = (const float*)d_in[17];
  const float* bpv0 = (const float*)d_in[18];
  const float* Wpv1 = (const float*)d_in[19];
  const float* bpv1 = (const float*)d_in[20];
  const float* Wfcv = (const float*)d_in[21];
  const float* bfcv = (const float*)d_in[22];

  dim3 grid(B_SZ / 64);   // 32 persistent blocks, 64 batch rows each
  dim3 block(128);        // 4 waves, 16 rows (one WMMA M-tile) per wave
  rnn_persistent_kernel<<<grid, block, SMEM_BYTES, stream>>>(
      x, Wxc, bxc, Whc, Wxv, bxv, Whv, Wscc, Wscv, Wsvv, Wsvc,
      Wpc0, bpc0, Wpc1, bpc1, Wfcc, bfcc, Wpv0, bpv0, Wpv1, bpv1, Wfcv, bfcv,
      (float*)d_out);
}
